// TT_GRU_39402029974128
// MI455X (gfx1250) — matrix operand order repaired and checked by reading, not compiled, not run on hardware
//
#include <hip/hip_runtime.h>
#include <hip/hip_bf16.h>

// ---------------------------------------------------------------------------
// TT-GRU for MI455X (gfx1250, wave32). bf16 WMMA 16x16x32 with f32 accum.
//   Phase 0: prep per-lane WMMA B-fragments (TT cores + recurrent kernel).
//   Phase 1: fused 4-stage TT contraction, one 8-wave WG per (t,b), all
//            stage permutations done via LDS scatter of WMMA D registers.
//   Phase 2: persistent 32-wave WG runs the 40-step GRU recurrence with
//            WMMA, weights streamed from L2, h kept f32 + bf16 shadow in LDS.
// ---------------------------------------------------------------------------

typedef __attribute__((ext_vector_type(16))) __bf16 v16bf;
typedef __attribute__((ext_vector_type(8)))  float  v8f;

union V16 { uint4 q[2]; v16bf v; };

__device__ __forceinline__ unsigned short f2bf(float f) {
  // native f32 -> bf16 (RNE) via hardware cvt
  __bf16 h = (__bf16)f;
  return __builtin_bit_cast(unsigned short, h);
}

__device__ __forceinline__ float hsig(float x) {
  return fminf(fmaxf(x * (1.0f / 6.0f) + 0.5f, 0.0f), 1.0f);
}

__device__ __forceinline__ v8f wmma_bf16(v16bf a, v16bf b, v8f c) {
  // D = A(16x32) * B(32x16) + C, f32 accumulate
  return __builtin_amdgcn_wmma_f32_16x16x32_bf16(false, a, false, b,
                                                 (short)0, c, false, false);
}

// A-fragment (16-bit A 16x32, ISA 7.12.2): lane<16 holds row=cl K[0:8)&K[16:24),
// lane>=16 holds row=cl K[8:16)&K[24:32). Two ds_read_b128 per lane.
__device__ __forceinline__ v16bf loadA_lds(const unsigned short* buf, int strideU,
                                           int row, int kbase, int lane) {
  int off = row * strideU + kbase + ((lane & 16) ? 8 : 0);
  V16 u;
  u.q[0] = *(const uint4*)(buf + off);
  u.q[1] = *(const uint4*)(buf + off + 16);
  return u.v;
}

// Pre-swizzled B fragment: 32 bytes per lane, contiguous -> 2x global_load_b128.
__device__ __forceinline__ v16bf loadB_glob(const unsigned short* __restrict__ frag,
                                            int tile, int lane) {
  const uint4* p = (const uint4*)(frag + (size_t)tile * 512 + lane * 16);
  V16 u;
  u.q[0] = p[0];
  u.q[1] = p[1];
  return u.v;
}

// ---------------------------------------------------------------------------
// Phase 0: build B fragments.
//   B layout (ISA 7.12.5 dense analogue): lane -> N = lane%16,
//   elements e: K = (lane/16)*8 + (e<8 ? e : e+8).
// TT tiles: 0 = core3(18x16), 1..3 = core2(80x16) K-tiles, 4..6 = core1(80x16)
// K-tiles, 7 = core0(32x12 padded to N16).
// rk tiles: index = kt*48 + ntg  (kt in [0,8), ntg in [0,48): cols ntg*16..).
// ---------------------------------------------------------------------------
__global__ void prep_frags_kernel(const float* __restrict__ kern,
                                  const float* __restrict__ rk,
                                  unsigned short* __restrict__ frag_tt,
                                  unsigned short* __restrict__ frag_rk) {
  int tile = blockIdx.x;
  int lane = threadIdx.x;          // 32 threads = 1 wave
  int N = lane & 15;
  int khalf = (lane >> 4) * 8;
  unsigned short vals[16];
  if (tile < 8) {
    int base, s0, s1, kt;
    if (tile == 0)      { base = 0;    s0 = 18; s1 = 16; kt = 0; }
    else if (tile < 4)  { base = 288;  s0 = 80; s1 = 16; kt = tile - 1; }
    else if (tile < 7)  { base = 1568; s0 = 80; s1 = 16; kt = tile - 4; }
    else                { base = 2848; s0 = 32; s1 = 12; kt = 0; }
#pragma unroll
    for (int e = 0; e < 16; ++e) {
      int K = kt * 32 + khalf + (e < 8 ? e : e + 8);
      float v = (K < s0 && N < s1) ? kern[base + K * s1 + N] : 0.0f;
      vals[e] = f2bf(v);
    }
#pragma unroll
    for (int e = 0; e < 16; ++e) frag_tt[(size_t)tile * 512 + lane * 16 + e] = vals[e];
  } else {
    int idx = tile - 8;
    int kt = idx / 48;
    int ntg = idx % 48;
#pragma unroll
    for (int e = 0; e < 16; ++e) {
      int K = kt * 32 + khalf + (e < 8 ? e : e + 8);
      vals[e] = f2bf(rk[(size_t)K * 768 + ntg * 16 + N]);
    }
#pragma unroll
    for (int e = 0; e < 16; ++e) frag_rk[(size_t)idx * 512 + lane * 16 + e] = vals[e];
  }
}

// ---------------------------------------------------------------------------
// Phase 1: fused TT contraction. Grid (40,32) = (t,b); 256 threads = 8 waves.
// Chain per (b,i0) slice (indices: rows x cols, K padded to mult. of 32):
//   S3[i1*20+i2][r3*4+o3]  = X[400x32(K=18)]           @ core3-tile
//   S2[o3*20+i1][r2*4+o2]  = A2[80x96(K=80)]           @ core2-tiles
//   S1[o2*4+o3][r1*4+o1]   = A1[16x96(K=80)]           @ core1-tiles
//   mx[o0*64+o1*16+o2*4+o3]= A0[64x32(K=(i0,r1))]      @ core0-tile
// Each stage's D registers are scattered to LDS directly in the next stage's
// A layout (the reference's reshape/transpose chain).
// ---------------------------------------------------------------------------
__global__ __launch_bounds__(256) void tt_contract_kernel(
    const float* __restrict__ x,
    const unsigned short* __restrict__ frag_tt,
    float* __restrict__ MX) {
  __shared__ __attribute__((aligned(16))) unsigned short Xbuf[400 * 32];   // 25.0 KB
  __shared__ __attribute__((aligned(16))) unsigned short A2buf[80 * 96];   // 15.0 KB
  __shared__ __attribute__((aligned(16))) unsigned short A1buf[16 * 96];   //  3.0 KB
  __shared__ __attribute__((aligned(16))) unsigned short A0buf[64 * 32];   //  4.0 KB

  const int t = blockIdx.x, b = blockIdx.y;
  const int tid = threadIdx.x;
  const int wave = tid >> 5, lane = tid & 31;
  const int cl = lane & 15;
  const int rsel = (lane >> 4) << 3;   // D-reg row offset: 0 or 8

  // B fragments kept in VGPRs for the waves that need them
  v16bf B3 = loadB_glob(frag_tt, 0, lane);
  v16bf B2_0 = {}, B2_1 = {}, B2_2 = {};
  v16bf B1_0 = {}, B1_1 = {}, B1_2 = {};
  v16bf B0v = {};
  if (wave < 5) {
    B2_0 = loadB_glob(frag_tt, 1, lane);
    B2_1 = loadB_glob(frag_tt, 2, lane);
    B2_2 = loadB_glob(frag_tt, 3, lane);
  }
  if (wave == 0) {
    B1_0 = loadB_glob(frag_tt, 4, lane);
    B1_1 = loadB_glob(frag_tt, 5, lane);
    B1_2 = loadB_glob(frag_tt, 6, lane);
  }
  if (wave < 4) B0v = loadB_glob(frag_tt, 7, lane);

  // zero constant K-pad regions ONCE (Xbuf k=18..31, A2 k=80..95, A1 k=80..95)
  for (int i = tid; i < 400 * 14; i += 256) {
    int r = i / 14, k = i - r * 14;
    Xbuf[r * 32 + 18 + k] = 0;
  }
  for (int i = tid; i < 80 * 16; i += 256) A2buf[(i >> 4) * 96 + 80 + (i & 15)] = 0;
  for (int i = tid; i < 16 * 16; i += 256) A1buf[(i >> 4) * 96 + 80 + (i & 15)] = 0;

  const float* xp = x + ((size_t)b * 40 + t) * 57600;

  // stage-3 tile body (wave-uniform m; EXEC all ones inside)
  auto stage3_tile = [&](int m) {
    v16bf a = loadA_lds(Xbuf, 32, 16 * m + cl, 0, lane);
    v8f c = {};
    c = wmma_bf16(a, B3, c);
    int r3 = cl >> 2, o3 = cl & 3;
#pragma unroll
    for (int g = 0; g < 8; ++g) {
      int R = 16 * m + g + rsel;        // = i1*20+i2
      int i1 = R / 20, i2 = R % 20;
      A2buf[(o3 * 20 + i1) * 96 + i2 * 4 + r3] = f2bf(c[g]);
    }
  };

  for (int i0 = 0; i0 < 8; ++i0) {
    // ---- load x slice (20*20*18 contiguous f32 -> bf16) ----
    const float* xs = xp + i0 * 7200;
    for (int j = tid; j < 7200; j += 256) {
      int r = j / 18, k = j - r * 18;
      Xbuf[r * 32 + k] = f2bf(xs[j]);
    }
    __syncthreads();

    // ---- stage 3: 25 M-tiles; tile 24 goes to wave 7 (wave 0 owns stage 1) --
    for (int m = wave; m < 24; m += 8) stage3_tile(m);
    if (wave == 7) stage3_tile(24);
    __syncthreads();

    // ---- stage 2: 5 M-tiles x 3 K-tiles on waves 0..4 ----
    if (wave < 5) {
      int row = 16 * wave + cl;
      v8f c = {};
      c = wmma_bf16(loadA_lds(A2buf, 96, row, 0, lane),  B2_0, c);
      c = wmma_bf16(loadA_lds(A2buf, 96, row, 32, lane), B2_1, c);
      c = wmma_bf16(loadA_lds(A2buf, 96, row, 64, lane), B2_2, c);
      int r2 = cl >> 2, o2 = cl & 3;
#pragma unroll
      for (int g = 0; g < 8; ++g) {
        int row2 = 16 * wave + g + rsel;   // = o3*20+i1
        int o3 = row2 / 20, i1 = row2 % 20;
        A1buf[(o2 * 4 + o3) * 96 + i1 * 4 + r2] = f2bf(c[g]);
      }
    }
    __syncthreads();

    // ---- stage 1: 1 M-tile x 3 K-tiles on wave 0 ----
    if (wave == 0) {
      v8f c = {};
      c = wmma_bf16(loadA_lds(A1buf, 96, cl, 0, lane),  B1_0, c);
      c = wmma_bf16(loadA_lds(A1buf, 96, cl, 32, lane), B1_1, c);
      c = wmma_bf16(loadA_lds(A1buf, 96, cl, 64, lane), B1_2, c);
      int r1 = cl >> 2, o1 = cl & 3;
#pragma unroll
      for (int g = 0; g < 8; ++g) {
        int row1 = g + rsel;               // = o2*4+o3
        A0buf[(o1 * 16 + row1) * 32 + i0 * 4 + r1] = f2bf(c[g]);
      }
    }
    __syncthreads();
  }

  // ---- stage 0: 4 M-tiles on waves 0..3, write mx ----
  if (wave < 4) {
    v8f c = {};
    c = wmma_bf16(loadA_lds(A0buf, 32, 16 * wave + cl, 0, lane), B0v, c);
    if (cl < 12) {                          // o0 in [0,12)
      float* mxp = MX + ((size_t)t * 32 + b) * 768;
#pragma unroll
      for (int g = 0; g < 8; ++g) {
        int row = 16 * wave + g + rsel;     // = o1*16+o2*4+o3
        mxp[cl * 64 + row] = c[g];
      }
    }
  }
}

// ---------------------------------------------------------------------------
// Phase 2: 40-step GRU recurrence. 1 block, 1024 threads = 32 waves.
//   hbuf : h in f32 (master copy)          [32][256]
//   buf2 : bf16 shadow; alternately holds h(bf16) for the z/r GEMM and
//          (r*h)(bf16) for the candidate GEMM (same [b][k] layout).
// ---------------------------------------------------------------------------
__global__ __launch_bounds__(1024) void gru_kernel(
    const float* __restrict__ MX,
    const unsigned short* __restrict__ frag_rk,
    const float* __restrict__ bias,
    const float* __restrict__ cls_w,
    const float* __restrict__ cls_b,
    float* __restrict__ out) {
  __shared__ __attribute__((aligned(16))) float          hbuf[32 * 256];  // 32 KB
  __shared__ __attribute__((aligned(16))) unsigned short buf2[32 * 256];  // 16 KB

  const int tid = threadIdx.x;
  const int w = tid >> 5, lane = tid & 31;
  const int cl = lane & 15;
  const int rsel = (lane >> 4) << 3;

  for (int i = tid; i < 8192; i += 1024) { hbuf[i] = 0.0f; buf2[i] = 0; }
  __syncthreads();

  const int col = 16 * w + cl;               // z/r column this lane produces
  const float bcol = bias[col];
  const float bh = (w < 16) ? bias[512 + col] : 0.0f;

  for (int t = 0; t < 40; ++t) {
    const float* MXt = MX + (size_t)t * 32 * 768;

    // ---- z/r GEMM: [32,256] @ [256,512], wave w -> cols 16w.. ----
    v8f c0 = {}, c1 = {};
#pragma unroll
    for (int kt = 0; kt < 8; ++kt) {
      v16bf bf = loadB_glob(frag_rk, kt * 48 + w, lane);
      v16bf a0 = loadA_lds(buf2, 256, cl, kt * 32, lane);        // b = 0..15
      v16bf a1 = loadA_lds(buf2, 256, 16 + cl, kt * 32, lane);   // b = 16..31
      c0 = wmma_bf16(a0, bf, c0);
      c1 = wmma_bf16(a1, bf, c1);
    }
    __syncthreads();   // all h(bf16) reads from buf2 done

    // ---- gates: z kept in regs (w<16), r*h written bf16 into buf2 (w>=16) --
    float zv0[8], zv1[8];
    if (w < 16) {
#pragma unroll
      for (int g = 0; g < 8; ++g) {
        int b0 = g + rsel, b1 = b0 + 16;
        zv0[g] = hsig(MXt[b0 * 768 + col] + bcol + c0[g]);
        zv1[g] = hsig(MXt[b1 * 768 + col] + bcol + c1[g]);
      }
    } else {
      int u = col - 256;
#pragma unroll
      for (int g = 0; g < 8; ++g) {
        int b0 = g + rsel, b1 = b0 + 16;
        float r0 = hsig(MXt[b0 * 768 + col] + bcol + c0[g]);
        float r1 = hsig(MXt[b1 * 768 + col] + bcol + c1[g]);
        buf2[b0 * 256 + u] = f2bf(r0 * hbuf[b0 * 256 + u]);
        buf2[b1 * 256 + u] = f2bf(r1 * hbuf[b1 * 256 + u]);
      }
    }
    __syncthreads();   // (r*h) ready

    // ---- candidate GEMM: [32,256] @ [256,256] on waves 0..15 ----
    v8f d0 = {}, d1 = {};
    if (w < 16) {
#pragma unroll
      for (int kt = 0; kt < 8; ++kt) {
        v16bf bf = loadB_glob(frag_rk, kt * 48 + 32 + w, lane);
        v16bf a0 = loadA_lds(buf2, 256, cl, kt * 32, lane);
        v16bf a1 = loadA_lds(buf2, 256, 16 + cl, kt * 32, lane);
        d0 = wmma_bf16(a0, bf, d0);
        d1 = wmma_bf16(a1, bf, d1);
      }
    }
    __syncthreads();   // (r*h) reads done; buf2 reusable as h(bf16)

    // ---- h update (waves 0..15 own cols 0..255) ----
    if (w < 16) {
      int u = col;
#pragma unroll
      for (int g = 0; g < 8; ++g) {
        int b0 = g + rsel, b1 = b0 + 16;
        float hh0 = tanhf(MXt[b0 * 768 + 512 + u] + bh + d0[g]);
        float hh1 = tanhf(MXt[b1 * 768 + 512 + u] + bh + d1[g]);
        float ho0 = hbuf[b0 * 256 + u];
        float ho1 = hbuf[b1 * 256 + u];
        float hn0 = zv0[g] * ho0 + (1.0f - zv0[g]) * hh0;
        float hn1 = zv1[g] * ho1 + (1.0f - zv1[g]) * hh1;
        hbuf[b0 * 256 + u] = hn0;
        hbuf[b1 * 256 + u] = hn1;
        buf2[b0 * 256 + u] = f2bf(hn0);
        buf2[b1 * 256 + u] = f2bf(hn1);
      }
    }
    __syncthreads();   // h ready for next step
  }

  // ---- classification head: out[b][j] = h[b] . cls_w[j] + cls_b[j] ----
  if (tid < 352) {
    int b = tid / 11, j = tid - b * 11;
    float acc = cls_b[j];
    for (int u = 0; u < 256; ++u)
      acc += hbuf[b * 256 + u] * cls_w[j * 256 + u];
    out[b * 11 + j] = acc;
  }
}

// ---------------------------------------------------------------------------
extern "C" void kernel_launch(void* const* d_in, const int* in_sizes, int n_in,
                              void* d_out, int out_size, void* d_ws, size_t ws_size,
                              hipStream_t stream) {
  const float* x    = (const float*)d_in[0];   // [32][40][57600]
  const float* kern = (const float*)d_in[1];   // [3232]
  const float* rk   = (const float*)d_in[2];   // [256][768]
  const float* bias = (const float*)d_in[3];   // [768]
  const float* cw   = (const float*)d_in[4];   // [11][256]
  const float* cb   = (const float*)d_in[5];   // [11]
  float* out = (float*)d_out;                  // [32][11]

  const size_t MX_BYTES = (size_t)40 * 32 * 768 * sizeof(float);     // 3.93 MB
  float* MX = (float*)d_ws;
  unsigned short* frag_tt = (unsigned short*)((char*)d_ws + MX_BYTES);
  unsigned short* frag_rk = frag_tt + 8 * 512;
  if (ws_size < MX_BYTES + (size_t)(8 + 384) * 1024) return;

  prep_frags_kernel<<<392, 32, 0, stream>>>(kern, rk, frag_tt, frag_rk);
  tt_contract_kernel<<<dim3(40, 32), 256, 0, stream>>>(x, frag_tt, MX);
  gru_kernel<<<1, 1024, 0, stream>>>(MX, frag_rk, bias, cw, cb, out);
}
